// Attention_1735166787635
// MI455X (gfx1250) — matrix operand order repaired and checked by reading, not compiled, not run on hardware
//
#include <hip/hip_runtime.h>

#define NX    768
#define NQKV  2304
#define SEQ   4096
#define BATCH 2
#define HEADS 12
#define HDIM  64

typedef __attribute__((ext_vector_type(16))) __bf16 v16bf;
typedef __attribute__((ext_vector_type(8)))  __bf16 v8bf;
typedef __attribute__((ext_vector_type(8)))  float  v8f;

__device__ __forceinline__ v8bf ld8(const __bf16* p) { return *(const v8bf*)p; }
__device__ __forceinline__ v16bf ld16(const __bf16* p) { return *(const v16bf*)p; }

__device__ __forceinline__ v16bf cat(v8bf lo, v8bf hi) {
    v16bf r;
#pragma unroll
    for (int j = 0; j < 8; ++j) { r[j] = lo[j]; r[8 + j] = hi[j]; }
    return r;
}

__device__ __forceinline__ v8f wmma_bf16(v16bf a, v16bf b, v8f c) {
    return __builtin_amdgcn_wmma_f32_16x16x32_bf16(false, a, false, b, (short)0, c,
                                                   false, false);
}

// gfx1250 async global->LDS copy (ASYNCcnt-tracked), per CDNA5 ISA.
// VDST = per-lane LDS byte address, VADDR = per-lane 64-bit global address.
__device__ __forceinline__ void async_ld_b128(unsigned lds_off, const __bf16* g) {
    asm volatile("global_load_async_to_lds_b128 %0, %1, off"
                 :: "v"(lds_off), "v"(g) : "memory");
}
__device__ __forceinline__ void wait_async0() {
    asm volatile("s_wait_asynccnt 0x0" ::: "memory");
}
// Low 32 bits of a generic LDS pointer are the LDS byte offset.
__device__ __forceinline__ unsigned lds_off32(const void* p) {
    return (unsigned)(size_t)p;
}

// ---------------------------------------------------------------------------
// fp32 -> bf16 conversion (grid-stride)
// ---------------------------------------------------------------------------
__global__ void f32_to_bf16(const float* __restrict__ in, __bf16* __restrict__ out, int n) {
    int i = blockIdx.x * blockDim.x + threadIdx.x;
    int stride = gridDim.x * blockDim.x;
    for (; i < n; i += stride) out[i] = (__bf16)in[i];
}

// ---------------------------------------------------------------------------
// QKV GEMM: [8192 x 768] bf16 @ [768 x 2304] bf16 + bias -> Q/K/V bf16.
// Block tile 128x64, 8 waves in 4(M)x2(N), each wave a 32x32 region
// (2x2 WMMA register blocking). A tile via async global->LDS.
// ---------------------------------------------------------------------------
__global__ __launch_bounds__(256)
void gemm_qkv(const __bf16* __restrict__ A, const __bf16* __restrict__ W,
              const float* __restrict__ bias,
              __bf16* __restrict__ Q, __bf16* __restrict__ K, __bf16* __restrict__ V) {
    __shared__ __bf16 As[128][32];  // [m][k]
    __shared__ __bf16 Bs[64][32];   // [n][k] (transposed B tile)
    const int tid  = threadIdx.x;
    const int lane = tid & 31;
    const int wave = tid >> 5;
    const int bn = blockIdx.x * 64;
    const int bm = blockIdx.y * 128;
    const int wm = wave & 3;         // M sub-block (32 rows)
    const int wn = wave >> 2;        // N sub-block (32 cols)
    v8f a00 = {}, a01 = {}, a10 = {}, a11 = {};
    const int rl  = lane & 15;
    const int hi  = lane >> 4;
    const int kb  = hi ? 8 : 0;
    const int kst = hi ? 16 : 0;
    const int r0 = wm * 32 + rl, r1 = r0 + 16;
    const int n0 = wn * 32 + rl, n1 = n0 + 16;

    // Per-thread A-tile chunks (two b128 each): 128 rows x 32 cols
    const int c0 = tid * 2, c1 = tid * 2 + 1;
    const int ar0 = c0 >> 2, ad0 = (c0 & 3) * 8;
    const int ar1 = c1 >> 2, ad1 = (c1 & 3) * 8;
    const unsigned l0 = lds_off32(&As[ar0][ad0]);
    const unsigned l1 = lds_off32(&As[ar1][ad1]);

    for (int kk = 0; kk < NX; kk += 32) {
        async_ld_b128(l0, A + (size_t)(bm + ar0) * NX + kk + ad0);
        async_ld_b128(l1, A + (size_t)(bm + ar1) * NX + kk + ad1);
        {   // B tile transposed: w[k][n] contiguous in n -> Bs[n][k]
            int k = tid >> 3, nn = (tid & 7) * 8;
            v8bf v = *(const v8bf*)(W + (size_t)(kk + k) * NQKV + bn + nn);
#pragma unroll
            for (int j = 0; j < 8; ++j) Bs[nn + j][k] = v[j];
        }
        if (kk + 32 < NX)
            __builtin_prefetch(W + (size_t)(kk + 32 + (tid >> 3)) * NQKV + bn, 0, 1);
        wait_async0();
        __syncthreads();
        v16bf fA0 = ld16(&As[r0][kst]);
        v16bf fA1 = ld16(&As[r1][kst]);
        v16bf fB0 = ld16(&Bs[n0][kst]);
        v16bf fB1 = ld16(&Bs[n1][kst]);
        a00 = wmma_bf16(fA0, fB0, a00);
        a01 = wmma_bf16(fA0, fB1, a01);
        a10 = wmma_bf16(fA1, fB0, a10);
        a11 = wmma_bf16(fA1, fB1, a11);
        __syncthreads();
    }
#pragma unroll
    for (int ti = 0; ti < 2; ++ti) {
#pragma unroll
        for (int tj = 0; tj < 2; ++tj) {
            v8f acc = ti ? (tj ? a11 : a10) : (tj ? a01 : a00);
            int col = bn + wn * 32 + tj * 16 + rl;
            float bv = bias[col];
#pragma unroll
            for (int i = 0; i < 8; ++i) {
                int row = bm + wm * 32 + ti * 16 + i + hi * 8;
                __bf16 o = (__bf16)(acc[i] + bv);
                if (col < NX)          Q[(size_t)row * NX + col] = o;
                else if (col < 2 * NX) K[(size_t)row * NX + col - NX] = o;
                else                   V[(size_t)row * NX + col - 2 * NX] = o;
            }
        }
    }
}

// ---------------------------------------------------------------------------
// Flash attention over [B,H,S,64] bf16 (plain-reshape layout). One WG = 128
// q rows (8 waves x 16), one (b,h). K tile via async global->LDS; V tile
// transposed manually; online softmax fp32.
// ---------------------------------------------------------------------------
__global__ __launch_bounds__(256)
void flash_attn(const __bf16* __restrict__ Qg, const __bf16* __restrict__ Kg,
                const __bf16* __restrict__ Vg, const float* __restrict__ amask,
                __bf16* __restrict__ Og) {
    __shared__ __bf16 Ks[32][64];       // [key][d]
    __shared__ __bf16 Vt[64][32];       // [d][key]
    __shared__ __bf16 Ps[8][16][32];    // per-wave P scratch
    const int tid  = threadIdx.x;
    const int lane = tid & 31;
    const int wave = tid >> 5;
    const int bh = blockIdx.y;
    const int b = bh / HEADS;
    const size_t base = (size_t)b * SEQ * NX + (size_t)(bh % HEADS) * SEQ * HDIM;
    const int qb = blockIdx.x * 128 + wave * 16;
    const int rl  = lane & 15;
    const int hi  = lane >> 4;
    const int kb  = hi ? 8 : 0;
    const int kst = hi ? 16 : 0;

    const __bf16* qrow = Qg + base + (size_t)(qb + rl) * HDIM;
    v16bf aQ0 = ld16(qrow + kst);
    v16bf aQ1 = ld16(qrow + 32 + kst);

    v8f o0 = {}, o1 = {}, o2 = {}, o3 = {};
    float rmax[8], rsum[8];
#pragma unroll
    for (int i = 0; i < 8; ++i) { rmax[i] = -1e30f; rsum[i] = 0.0f; }

    const int key = tid >> 3, d0 = (tid & 7) * 8;
    const unsigned lK = lds_off32(&Ks[key][d0]);

    const int ntiles = blockIdx.x * 4 + 4;
    for (int kt = 0; kt < ntiles; ++kt) {
        size_t roff = base + (size_t)(kt * 32 + key) * HDIM + d0;
        async_ld_b128(lK, Kg + roff);               // K tile: direct async copy
        {                                           // V tile: load + transpose
            v8bf vv = *(const v8bf*)(Vg + roff);
#pragma unroll
            for (int j = 0; j < 8; ++j) Vt[d0 + j][key] = vv[j];
        }
        wait_async0();
        __syncthreads();
        if (kt * 32 <= qb + 15) {
            v8f s0 = {}, s1 = {};
            {
                v16bf bK0 = ld16(&Ks[rl][kst]);
                v16bf bK1 = ld16(&Ks[rl][32 + kst]);
                s0 = wmma_bf16(aQ0, bK0, s0);
                s0 = wmma_bf16(aQ1, bK1, s0);
                v16bf cK0 = ld16(&Ks[16 + rl][kst]);
                v16bf cK1 = ld16(&Ks[16 + rl][32 + kst]);
                s1 = wmma_bf16(aQ0, cK0, s1);
                s1 = wmma_bf16(aQ1, cK1, s1);
            }
            const int k0 = kt * 32 + rl;
            const int k1 = kt * 32 + 16 + rl;
            const float am0 = amask[b * SEQ + k0];
            const float am1 = amask[b * SEQ + k1];
#pragma unroll
            for (int i = 0; i < 8; ++i) {
                int row = qb + i + hi * 8;
                float v0 = (k0 <= row) ? (s0[i] * 0.125f + am0) : (-10000.0f + am0);
                float v1 = (k1 <= row) ? (s1[i] * 0.125f + am1) : (-10000.0f + am1);
                float tm = fmaxf(v0, v1);
                tm = fmaxf(tm, __shfl_xor(tm, 1, 32));
                tm = fmaxf(tm, __shfl_xor(tm, 2, 32));
                tm = fmaxf(tm, __shfl_xor(tm, 4, 32));
                tm = fmaxf(tm, __shfl_xor(tm, 8, 32));
                float nm   = fmaxf(rmax[i], tm);
                float corr = __expf(rmax[i] - nm);
                float e0 = __expf(v0 - nm);
                float e1 = __expf(v1 - nm);
                float ts = e0 + e1;
                ts += __shfl_xor(ts, 1, 32);
                ts += __shfl_xor(ts, 2, 32);
                ts += __shfl_xor(ts, 4, 32);
                ts += __shfl_xor(ts, 8, 32);
                rsum[i] = rsum[i] * corr + ts;
                rmax[i] = nm;
                o0[i] *= corr; o1[i] *= corr; o2[i] *= corr; o3[i] *= corr;
                int prow = i + hi * 8;
                Ps[wave][prow][rl]      = (__bf16)e0;
                Ps[wave][prow][16 + rl] = (__bf16)e1;
            }
            v16bf aP  = ld16(&Ps[wave][rl][kst]);
            v16bf bV0 = ld16(&Vt[0 * 16 + rl][kst]);
            v16bf bV1 = ld16(&Vt[1 * 16 + rl][kst]);
            v16bf bV2 = ld16(&Vt[2 * 16 + rl][kst]);
            v16bf bV3 = ld16(&Vt[3 * 16 + rl][kst]);
            o0 = wmma_bf16(aP, bV0, o0);
            o1 = wmma_bf16(aP, bV1, o1);
            o2 = wmma_bf16(aP, bV2, o2);
            o3 = wmma_bf16(aP, bV3, o3);
        }
        __syncthreads();
    }
#pragma unroll
    for (int i = 0; i < 8; ++i) {
        int row = qb + i + hi * 8;
        float inv = 1.0f / rsum[i];
        size_t ro = base + (size_t)row * HDIM;
        Og[ro + 0 * 16 + rl] = (__bf16)(o0[i] * inv);
        Og[ro + 1 * 16 + rl] = (__bf16)(o1[i] * inv);
        Og[ro + 2 * 16 + rl] = (__bf16)(o2[i] * inv);
        Og[ro + 3 * 16 + rl] = (__bf16)(o3[i] * inv);
    }
}

// ---------------------------------------------------------------------------
// Output projection: [8192 x 768] @ w_proj [768 x 768] + b_proj -> fp32 out.
// Same 128x64 / 2x2 blocking; head transpose folded into async A addresses.
// ---------------------------------------------------------------------------
__global__ __launch_bounds__(256)
void gemm_proj(const __bf16* __restrict__ Aatt, const __bf16* __restrict__ W,
               const float* __restrict__ bias, float* __restrict__ out) {
    __shared__ __bf16 As[128][32];
    __shared__ __bf16 Bs[64][32];
    const int tid  = threadIdx.x;
    const int lane = tid & 31;
    const int wave = tid >> 5;
    const int bn = blockIdx.x * 64;
    const int bm = blockIdx.y * 128;
    const int wm = wave & 3;
    const int wn = wave >> 2;
    v8f a00 = {}, a01 = {}, a10 = {}, a11 = {};
    const int rl  = lane & 15;
    const int hi  = lane >> 4;
    const int kb  = hi ? 8 : 0;
    const int kst = hi ? 16 : 0;
    const int r0 = wm * 32 + rl, r1 = r0 + 16;
    const int n0 = wn * 32 + rl, n1 = n0 + 16;

    const int c0 = tid * 2, c1 = tid * 2 + 1;
    const int ar0 = c0 >> 2, ad0 = (c0 & 3) * 8;
    const int ar1 = c1 >> 2, ad1 = (c1 & 3) * 8;
    const unsigned l0 = lds_off32(&As[ar0][ad0]);
    const unsigned l1 = lds_off32(&As[ar1][ad1]);
    const int gm0 = bm + ar0, gm1 = bm + ar1;
    const size_t rb0 = (size_t)(gm0 >> 12) * SEQ * NX + (size_t)(gm0 & (SEQ - 1)) * HDIM;
    const size_t rb1 = (size_t)(gm1 >> 12) * SEQ * NX + (size_t)(gm1 & (SEQ - 1)) * HDIM;

    for (int kk = 0; kk < NX; kk += 32) {
        {   // A tile with channel->(h,d) transpose, async copies
            int k0c = kk + ad0, k1c = kk + ad1;
            async_ld_b128(l0, Aatt + rb0 + (size_t)(k0c >> 6) * SEQ * HDIM + (k0c & 63));
            async_ld_b128(l1, Aatt + rb1 + (size_t)(k1c >> 6) * SEQ * HDIM + (k1c & 63));
        }
        {
            int k = tid >> 3, nn = (tid & 7) * 8;
            v8bf v = *(const v8bf*)(W + (size_t)(kk + k) * NX + bn + nn);
#pragma unroll
            for (int j = 0; j < 8; ++j) Bs[nn + j][k] = v[j];
        }
        if (kk + 32 < NX)
            __builtin_prefetch(W + (size_t)(kk + 32 + (tid >> 3)) * NX + bn, 0, 1);
        wait_async0();
        __syncthreads();
        v16bf fA0 = ld16(&As[r0][kst]);
        v16bf fA1 = ld16(&As[r1][kst]);
        v16bf fB0 = ld16(&Bs[n0][kst]);
        v16bf fB1 = ld16(&Bs[n1][kst]);
        a00 = wmma_bf16(fA0, fB0, a00);
        a01 = wmma_bf16(fA0, fB1, a01);
        a10 = wmma_bf16(fA1, fB0, a10);
        a11 = wmma_bf16(fA1, fB1, a11);
        __syncthreads();
    }
#pragma unroll
    for (int ti = 0; ti < 2; ++ti) {
#pragma unroll
        for (int tj = 0; tj < 2; ++tj) {
            v8f acc = ti ? (tj ? a11 : a10) : (tj ? a01 : a00);
            int col = bn + wn * 32 + tj * 16 + rl;
            float bv = bias[col];
#pragma unroll
            for (int i = 0; i < 8; ++i) {
                int row = bm + wm * 32 + ti * 16 + i + hi * 8;
                out[(size_t)row * NX + col] = acc[i] + bv;
            }
        }
    }
}

// ---------------------------------------------------------------------------
extern "C" void kernel_launch(void* const* d_in, const int* in_sizes, int n_in,
                              void* d_out, int out_size, void* d_ws, size_t ws_size,
                              hipStream_t stream) {
    const float* hidden = (const float*)d_in[0];
    const float* amask  = (const float*)d_in[1];
    const float* w_attn = (const float*)d_in[2];
    const float* b_attn = (const float*)d_in[3];
    const float* w_proj = (const float*)d_in[4];
    const float* b_proj = (const float*)d_in[5];
    float* out = (float*)d_out;

    __bf16* ws = (__bf16*)d_ws;
    const size_t nHid = (size_t)BATCH * SEQ * NX;
    __bf16* hidBF  = ws;
    __bf16* wAtnBF = hidBF + nHid;
    __bf16* wPrjBF = wAtnBF + (size_t)NX * NQKV;
    __bf16* Qbf    = wPrjBF + (size_t)NX * NX;
    __bf16* Kbf    = Qbf + nHid;
    __bf16* Vbf    = Kbf + nHid;
    __bf16* Obf    = Vbf + nHid;

    f32_to_bf16<<<2048, 256, 0, stream>>>(hidden, hidBF, (int)nHid);
    f32_to_bf16<<<1024, 256, 0, stream>>>(w_attn, wAtnBF, NX * NQKV);
    f32_to_bf16<<<512, 256, 0, stream>>>(w_proj, wPrjBF, NX * NX);

    gemm_qkv<<<dim3(NQKV / 64, (BATCH * SEQ) / 128), 256, 0, stream>>>(
        hidBF, wAtnBF, b_attn, Qbf, Kbf, Vbf);

    flash_attn<<<dim3(SEQ / 128, BATCH * HEADS), 256, 0, stream>>>(
        Qbf, Kbf, Vbf, amask, Obf);

    gemm_proj<<<dim3(NX / 64, (BATCH * SEQ) / 128), 256, 0, stream>>>(
        Obf, wPrjBF, b_proj, out);
}
